// Head_38878043964021
// MI455X (gfx1250) — matrix operand order repaired and checked by reading, not compiled, not run on hardware
//
#include <hip/hip_runtime.h>
#include <stdint.h>

typedef __attribute__((ext_vector_type(16))) _Float16     v16h;
typedef __attribute__((ext_vector_type(8)))  _Float16     v8h;
typedef __attribute__((ext_vector_type(8)))  float        v8f;
typedef __attribute__((ext_vector_type(4)))  unsigned int u32x4;
typedef __attribute__((ext_vector_type(8)))  int          i32x8;
typedef __attribute__((ext_vector_type(4)))  int          i32x4;
typedef __attribute__((ext_vector_type(4)))  float        f32x4;

#define B_    16
#define T_    2048
#define D_    128
#define ROW_  384            // 3*D (k|q|v packed)
#define BQ_   64             // query rows per workgroup (4 waves x 16)
#define BK_   32             // keys per inner step
#define LROW_ 132            // raw f32 LDS row stride (staged path)
#define KHS_  136            // K f16 LDS row stride in halves (272B: aligned, bank-shift 4)
#define VTS_  40             // V^T f16 LDS row stride in halves (80B: aligned, bank-shift 20)
#define PS_   40             // P f16 row stride in halves

// fast-path (PRE) LDS layout: double-buffered f16 tiles
enum : unsigned { P_KH0 = 0, P_KH1 = 8704, P_VT0 = 17408, P_VT1 = 27648, P_PB = 37888, P_SZ = 43008 };
// staged-path LDS layout: raw f32 double buffers + f16 staging
enum : unsigned { S_RAWK0 = 0, S_RAWV0 = 16896, S_RAWK1 = 33792, S_RAWV1 = 50688,
                  S_KH = 67584, S_VT = 76288, S_PB = 86528, S_SZ = 91648 };

// TDM group1 word0 variants: data_size | pad_enable | pad_interval | pad_amount(4 dwords)
#define W0_F32 ((2 << 16) | (1 << 20) | (6 << 22) | (3 << 25))   // 4B elems, pad every 128 dwords
#define W0_K16 ((1 << 16) | (1 << 20) | (5 << 22) | (3 << 25))   // 2B elems, pad every 64 dwords
#define W0_V16 ((1 << 16) | (1 << 20) | (3 << 22) | (3 << 25))   // 2B elems, pad every 16 dwords

#if __has_builtin(__builtin_amdgcn_tensor_load_to_lds)
#define USE_TDM 1
#else
#define USE_TDM 0
#endif

__device__ __forceinline__ v16h cat8(v8h a, v8h b)
{
    return __builtin_shufflevector(a, b, 0,1,2,3,4,5,6,7,8,9,10,11,12,13,14,15);
}

// max over the 16-lane DPP row (wave32: two independent 16-lane groups) on the
// VALU pipe — rotate-combine butterfly, no LDS/bpermute traffic.
__device__ __forceinline__ float rowmax16(float v)
{
    v = fmaxf(v, __int_as_float(__builtin_amdgcn_update_dpp(0, __float_as_int(v), 0x128, 0xf, 0xf, true))); // row_ror:8
    v = fmaxf(v, __int_as_float(__builtin_amdgcn_update_dpp(0, __float_as_int(v), 0x124, 0xf, 0xf, true))); // row_ror:4
    v = fmaxf(v, __int_as_float(__builtin_amdgcn_update_dpp(0, __float_as_int(v), 0x122, 0xf, 0xf, true))); // row_ror:2
    v = fmaxf(v, __int_as_float(__builtin_amdgcn_update_dpp(0, __float_as_int(v), 0x121, 0xf, 0xf, true))); // row_ror:1
    return v;
}

#if USE_TDM
// Generic 2D TDM tile load: tile1 rows x tile0 elems, global row stride stride0.
__device__ __forceinline__ void tdm_load(const void* g, unsigned lds_off, int w0,
                                         unsigned dim0, unsigned dim1,
                                         unsigned tile0, unsigned tile1, unsigned stride0)
{
    uint64_t ga = (uint64_t)(uintptr_t)g;
    u32x4 d0;
    d0[0] = 1u;                                                   // count=1 (valid user D#)
    d0[1] = lds_off;                                              // lds_addr
    d0[2] = (unsigned)ga;                                         // global_addr[31:0]
    d0[3] = (unsigned)((ga >> 32) & 0x01ffffffu) | 0x80000000u;   // global_addr[56:32] | type=2
    i32x8 d1;
    d1[0] = w0;
    d1[1] = (int)(dim0 << 16);                                    // tensor_dim0 (bits 79:48)
    d1[2] = (int)((dim0 >> 16) | (dim1 << 16));                   // tensor_dim1 (bits 111:80)
    d1[3] = (int)((dim1 >> 16) | (tile0 << 16));                  // tile_dim0 (bits 127:112)
    d1[4] = (int)tile1;                                           // tile_dim1, tile_dim2=0
    d1[5] = (int)stride0;                                         // tensor_dim0_stride
    d1[6] = 0;
    d1[7] = 0;
    i32x4 z4 = {0, 0, 0, 0};
#if defined(__clang_major__) && (__clang_major__ >= 23)
    i32x8 z8 = {0, 0, 0, 0, 0, 0, 0, 0};
    __builtin_amdgcn_tensor_load_to_lds(d0, d1, z4, z4, z8, 0);
#else
    __builtin_amdgcn_tensor_load_to_lds(d0, d1, z4, z4, 0);
#endif
}
#endif

// ---- pre-pack: x(f32) -> Qh (scaled f16), Kh (f16), V^T (f16, [B][128][T])
__global__ __launch_bounds__(128, 1)
void qkv_prepack(const float* __restrict__ x, _Float16* __restrict__ qh,
                 _Float16* __restrict__ kh, _Float16* __restrict__ vt)
{
    __shared__ _Float16 lv[128][130];
    const int b   = blockIdx.y;
    const int t0  = blockIdx.x * 128;
    const int tid = threadIdx.x;
    const float qs = 0.08838834764831845f * 1.4426950408889634f;  // rsqrt(128)*log2(e)

    for (int r = 0; r < 128; ++r) {
        const size_t grow = (size_t)b * T_ + t0 + r;
        const float* xr = x + grow * ROW_;
        kh[grow * D_ + tid] = (_Float16)xr[tid];
        qh[grow * D_ + tid] = (_Float16)(xr[D_ + tid] * qs);
        lv[r][tid] = (_Float16)xr[2 * D_ + tid];
    }
    __syncthreads();
    const int g = tid >> 6, s2 = tid & 63;   // transpose write, coalesced along T
    for (int cl = 0; cl < 64; ++cl) {
        const int c = cl + g * 64;
        union { _Float16 h[2]; unsigned u; } pk;
        pk.h[0] = lv[2 * s2][c];
        pk.h[1] = lv[2 * s2 + 1][c];
        *(unsigned*)(vt + ((size_t)b * D_ + c) * T_ + t0 + 2 * s2) = pk.u;
    }
}

template<bool PRE>
__global__ __launch_bounds__(128, 1)
void fa_fwd(const float* __restrict__ x, const _Float16* __restrict__ qh,
            const _Float16* __restrict__ khg, const _Float16* __restrict__ vtg,
            float* __restrict__ out)
{
    __shared__ __align__(16) unsigned char smem[PRE ? P_SZ : S_SZ];

    const int b    = blockIdx.y;
    const int qb   = blockIdx.x;
    const int tid  = threadIdx.x;
    const int wave = tid >> 5;
    const int lane = tid & 31;
    const int l16  = lane & 15;
    const int half = lane >> 4;

    const int ntiles = qb * 2 + 2;                // causal: keys 0 .. qb*64+63 only
    const int qrow0  = qb * BQ_ + wave * 16;

#if USE_TDM
    if (wave == 0) {   // prologue: kick tile 0 DMA, overlaps Q fragment load
        if (PRE) {
            tdm_load(khg + (size_t)(b * T_) * D_, P_KH0, W0_K16, 128u, 32768u, 128u, 32u, 128u);
            tdm_load(vtg + (size_t)b * D_ * T_,   P_VT0, W0_V16, 2048u, 2048u, 32u, 128u, 2048u);
        } else {
            const float* kg0 = x + (size_t)(b * T_) * ROW_;
            tdm_load(kg0,          S_RAWK0, W0_F32, 384u, 32768u, 128u, 32u, 384u);
            tdm_load(kg0 + 2 * D_, S_RAWV0, W0_F32, 384u, 32768u, 128u, 32u, 384u);
        }
    }
#endif

    const float qs = 0.08838834764831845f * 1.4426950408889634f;

    // ---- Q fragments: four 16x32 f16 A-operands (lane = row, K split per half)
    v16h aq[4];
    if (PRE) {
        const _Float16* q = qh + (size_t)(b * T_ + qrow0 + l16) * D_;
#pragma unroll
        for (int c = 0; c < 4; ++c) {
            const int k0 = c * 32 + half * 8;
            aq[c] = cat8(*(const v8h*)(q + k0), *(const v8h*)(q + k0 + 16));
        }
    } else {
        const float* q = x + ((size_t)(b * T_ + qrow0 + l16)) * ROW_ + D_;
#pragma unroll
        for (int c = 0; c < 4; ++c) {
            const int k0 = c * 32 + half * 8;
            f32x4 a0 = *(const f32x4*)(q + k0);
            f32x4 a1 = *(const f32x4*)(q + k0 + 4);
            f32x4 a2 = *(const f32x4*)(q + k0 + 16);
            f32x4 a3 = *(const f32x4*)(q + k0 + 20);
#pragma unroll
            for (int e = 0; e < 4; ++e) {
                aq[c][e]      = (_Float16)(a0[e] * qs);
                aq[c][4 + e]  = (_Float16)(a1[e] * qs);
                aq[c][8 + e]  = (_Float16)(a2[e] * qs);
                aq[c][12 + e] = (_Float16)(a3[e] * qs);
            }
        }
    }

    v16h vones;
#pragma unroll
    for (int j = 0; j < 16; ++j) vones[j] = (_Float16)1.0f;

    v8f o[8] = {};
    v8f ol = {};                                  // softmax denominator tile: ol = P_cum @ 1
    float mrow[8];
#pragma unroll
    for (int i = 0; i < 8; ++i) mrow[i] = -__builtin_huge_valf();

    _Float16* pbuf = (_Float16*)(smem + (PRE ? P_PB : S_PB)) + wave * (16 * PS_);

    for (int kt = 0; kt < ntiles; ++kt) {
        const int krow0 = kt * BK_;

        if (PRE) {
#if USE_TDM
            __syncthreads();   // WAR: all waves done reading buf[(kt+1)&1]
            if (wave == 0) {
                if (kt + 1 < ntiles) {
                    const int kr = krow0 + BK_;
                    const unsigned kof = ((kt + 1) & 1) ? P_KH1 : P_KH0;
                    const unsigned vof = ((kt + 1) & 1) ? P_VT1 : P_VT0;
                    tdm_load(khg + (size_t)(b * T_ + kr) * D_, kof, W0_K16, 128u, 32768u, 128u, 32u, 128u);
                    tdm_load(vtg + (size_t)b * D_ * T_ + kr,   vof, W0_V16, 2048u, 2048u, 32u, 128u, 2048u);
                    __builtin_amdgcn_s_wait_tensorcnt(2);   // in-order TDM: tile kt landed
                } else {
                    __builtin_amdgcn_s_wait_tensorcnt(0);
                }
            }
            __syncthreads();   // tile kt visible
#endif
        } else {
#if USE_TDM
            if (wave == 0) {
                if (kt + 1 < ntiles) {
                    const float* kg = x + ((size_t)(b * T_ + (kt + 1) * BK_)) * ROW_;
                    const unsigned ko = ((kt + 1) & 1) ? S_RAWK1 : S_RAWK0;
                    const unsigned vo = ((kt + 1) & 1) ? S_RAWV1 : S_RAWV0;
                    tdm_load(kg,          ko, W0_F32, 384u, 32768u, 128u, 32u, 384u);
                    tdm_load(kg + 2 * D_, vo, W0_F32, 384u, 32768u, 128u, 32u, 384u);
                    __builtin_amdgcn_s_wait_tensorcnt(2);
                } else {
                    __builtin_amdgcn_s_wait_tensorcnt(0);
                }
            }
#endif
            __syncthreads();
            // cooperative f32 -> f16 staging
#if USE_TDM
            const float* srcK = (const float*)(smem + ((kt & 1) ? S_RAWK1 : S_RAWK0));
            const float* srcV = (const float*)(smem + ((kt & 1) ? S_RAWV1 : S_RAWV0));
            const int sstr = LROW_;
#else
            const float* srcK = x + ((size_t)(b * T_ + kt * BK_)) * ROW_;
            const float* srcV = srcK + 2 * D_;
            const int sstr = ROW_;
#endif
            {
                const int r  = tid & 31;
                const int c0 = (tid >> 5) * 32;
                const float* s = srcK + (size_t)r * sstr + c0;
                _Float16*    d = (_Float16*)(smem + S_KH) + r * KHS_ + c0;
#pragma unroll
                for (int j = 0; j < 32; j += 8) {
                    f32x4 a = *(const f32x4*)(s + j);
                    f32x4 c = *(const f32x4*)(s + j + 4);
                    v8h h;
#pragma unroll
                    for (int e = 0; e < 4; ++e) { h[e] = (_Float16)a[e]; h[4 + e] = (_Float16)c[e]; }
                    *(v8h*)(d + j) = h;
                }
            }
            {
                const float* s = srcV + tid;
                _Float16*    d = (_Float16*)(smem + S_VT) + tid * VTS_;
#pragma unroll
                for (int k = 0; k < 32; k += 8) {
                    v8h h;
#pragma unroll
                    for (int e = 0; e < 8; ++e) h[e] = (_Float16)s[(size_t)(k + e) * sstr];
                    *(v8h*)(d + k) = h;
                }
            }
            __syncthreads();
        }

        const _Float16* Kh = PRE ? (const _Float16*)(smem + (((kt & 1) ? P_KH1 : P_KH0)))
                                 : (const _Float16*)(smem + S_KH);
        const _Float16* Vt = PRE ? (const _Float16*)(smem + (((kt & 1) ? P_VT1 : P_VT0)))
                                 : (const _Float16*)(smem + S_VT);

        // ---- S = Q @ K^T : preload all B-fragments, then WMMA chains
        v16h bk0[4], bk1[4];
#pragma unroll
        for (int c = 0; c < 4; ++c) {
            const v8h* k0p = (const v8h*)(Kh + (size_t)l16 * KHS_ + c * 32 + half * 8);
            const v8h* k1p = (const v8h*)(Kh + (size_t)(16 + l16) * KHS_ + c * 32 + half * 8);
            bk0[c] = cat8(k0p[0], k0p[2]);
            bk1[c] = cat8(k1p[0], k1p[2]);
        }
        v8f s0 = {}, s1 = {};
#pragma unroll
        for (int c = 0; c < 4; ++c)
            s0 = __builtin_amdgcn_wmma_f32_16x16x32_f16(false, aq[c], false, bk0[c], (short)0, s0, false, false);
#pragma unroll
        for (int c = 0; c < 4; ++c)
            s1 = __builtin_amdgcn_wmma_f32_16x16x32_f16(false, aq[c], false, bk1[c], (short)0, s1, false, false);

        // ---- causal mask only on tiles touching the diagonal (wave-uniform test)
        if (krow0 + BK_ - 1 > qrow0) {
#pragma unroll
            for (int i = 0; i < 8; ++i) {
                const int row = qrow0 + i + half * 8;
                const int c0  = krow0 + l16;
                s0[i] = (c0 > row)      ? -__builtin_huge_valf() : s0[i];
                s1[i] = (c0 + 16 > row) ? -__builtin_huge_valf() : s1[i];
            }
        }

        // ---- online softmax (exp2 domain); row max via DPP16 rotations (VALU only)
#pragma unroll
        for (int i = 0; i < 8; ++i) {
            const float v    = rowmax16(fmaxf(s0[i], s1[i]));
            const float mnew = fmaxf(mrow[i], v);
            const float corr = exp2f(mrow[i] - mnew);
            mrow[i] = mnew;
            const float p0 = exp2f(s0[i] - mnew);
            const float p1 = exp2f(s1[i] - mnew);
            ol[i] *= corr;
#pragma unroll
            for (int n = 0; n < 8; ++n) o[n][i] *= corr;
            pbuf[(i + half * 8) * PS_ + l16]      = (_Float16)p0;
            pbuf[(i + half * 8) * PS_ + l16 + 16] = (_Float16)p1;
        }

        // ---- reload P as A operand (same-wave DS ops are ordered): two b128s
        v16h ap;
        {
            const _Float16* pr = pbuf + l16 * PS_ + half * 8;
            ap = cat8(*(const v8h*)pr, *(const v8h*)(pr + 16));
        }

        // ---- O += P @ V ; denominator fused as extra WMMA: ol += P @ 1
        v16h bv[8];
#pragma unroll
        for (int n = 0; n < 8; ++n) {
            const v8h* vp = (const v8h*)(Vt + (size_t)(n * 16 + l16) * VTS_ + half * 8);
            bv[n] = cat8(vp[0], vp[2]);
        }
#pragma unroll
        for (int n = 0; n < 8; ++n)
            o[n] = __builtin_amdgcn_wmma_f32_16x16x32_f16(false, ap, false, bv[n], (short)0, o[n], false, false);
        ol = __builtin_amdgcn_wmma_f32_16x16x32_f16(false, ap, false, vones, (short)0, ol, false, false);
    }

    // ---- epilogue: normalize by l (row sums in ol) and store coalesced
#pragma unroll
    for (int i = 0; i < 8; ++i) {
        const int   row = qrow0 + i + half * 8;
        const float inv = 1.0f / ol[i];
        float* op = out + ((size_t)(b * T_ + row)) * D_ + l16;
#pragma unroll
        for (int n = 0; n < 8; ++n) op[n * 16] = o[n][i] * inv;
    }
}

extern "C" void kernel_launch(void* const* d_in, const int* in_sizes, int n_in,
                              void* d_out, int out_size, void* d_ws, size_t ws_size,
                              hipStream_t stream)
{
    (void)in_sizes; (void)n_in; (void)out_size;
    const float* x = (const float*)d_in[0];
    float* out     = (float*)d_out;
    dim3 grid(T_ / BQ_, B_);
    dim3 block(128);

#if USE_TDM
    const size_t npack = (size_t)B_ * T_ * D_;                 // elems per tensor
    if (d_ws != nullptr && ws_size >= 3 * npack * sizeof(_Float16)) {
        _Float16* qh = (_Float16*)d_ws;
        _Float16* kh = qh + npack;
        _Float16* vt = kh + npack;
        hipLaunchKernelGGL(qkv_prepack, dim3(T_ / 128, B_), dim3(128), 0, stream, x, qh, kh, vt);
        hipLaunchKernelGGL((fa_fwd<true>), grid, block, 0, stream, x, qh, kh, vt, out);
        return;
    }
#endif
    (void)ws_size;
    hipLaunchKernelGGL((fa_fwd<false>), grid, block, 0, stream,
                       x, (const _Float16*)nullptr, (const _Float16*)nullptr, (const _Float16*)nullptr, out);
}
